// DynamicPlasticityModule_51857435132577
// MI455X (gfx1250) — matrix operand order repaired and checked by reading, not compiled, not run on hardware
//
#include <hip/hip_runtime.h>
#include <hip/hip_bf16.h>

// ---------------------------------------------------------------------------
// Types for WMMA
// ---------------------------------------------------------------------------
typedef __bf16 bf16x16 __attribute__((ext_vector_type(16)));
typedef float  f32x8   __attribute__((ext_vector_type(8)));

union FragAB {
    bf16x16 v;
    uint4   u[2];
};

__device__ __forceinline__ unsigned short f32_to_bf16_rne(float f) {
    unsigned int u = __float_as_uint(f);
    unsigned int rounding = 0x7FFFu + ((u >> 16) & 1u);
    u += rounding;
    return (unsigned short)(u >> 16);
}

__device__ __forceinline__ float gelu_tanh(float v) {
    const float k0 = 0.7978845608028654f; // sqrt(2/pi)
    float inner = k0 * (v + 0.044715f * v * v * v);
    return 0.5f * v * (1.0f + tanhf(inner));
}

// gfx1250 async global->LDS copy (ASYNCcnt-tracked, no VGPR round trip).
// lds_off = byte offset into the workgroup's LDS allocation (low 32 bits of a
// generic pointer to a __shared__ object), gaddr = 64-bit global address.
__device__ __forceinline__ void async_load_b128(unsigned lds_off,
                                                const void* gaddr) {
    asm volatile("global_load_async_to_lds_b128 %0, %1, off"
                 :: "v"(lds_off), "v"((unsigned long long)gaddr)
                 : "memory");
}

// ---------------------------------------------------------------------------
// Problem constants
// ---------------------------------------------------------------------------
#define BB    8192
#define DIN   1024
#define DOUT  1024
#define NEXP  8
#define HID   64

#define TILE_M 128
#define TILE_N 128
#define TILE_K 64          // two 16x16x32 WMMA K-steps per staged tile
#define LDS_STRIDE 72      // padded row stride (elements); 144B rows, 16B aligned

// ---------------------------------------------------------------------------
// Kernel 1: input spikes + x -> bf16
// softmax(scale_weights) always sums to 1 => membrane == x => spike = x >= 0.8
// ---------------------------------------------------------------------------
__global__ void __launch_bounds__(256)
prep_kernel(const float* __restrict__ x, unsigned short* __restrict__ xbf,
            float* __restrict__ spikes, int n) {
    int i = blockIdx.x * 256 + threadIdx.x;
    if (i < n) {
        float v = x[i];
        spikes[i] = (v >= 0.8f) ? 1.0f : 0.0f;
        xbf[i] = f32_to_bf16_rne(v);
    }
}

// nano_W -> bf16
__global__ void __launch_bounds__(256)
convw_kernel(const float* __restrict__ w, unsigned short* __restrict__ wbf, int n) {
    int i = blockIdx.x * 256 + threadIdx.x;
    if (i < n) wbf[i] = f32_to_bf16_rne(w[i]);
}

// ---------------------------------------------------------------------------
// Kernel 2: router. One 256-thread (8-wave) workgroup per row of x.
// hbuf[0..7] = gate scores, hbuf[8..71] = capacity-net hidden pre-LN.
// ---------------------------------------------------------------------------
__global__ void __launch_bounds__(256)
router_kernel(const float* __restrict__ x,      const float* __restrict__ gateW,
              const float* __restrict__ capW1,  const float* __restrict__ capB1,
              const float* __restrict__ lnG,    const float* __restrict__ lnB,
              const float* __restrict__ capW2,  const float* __restrict__ capB2,
              const float* __restrict__ temperature,
              float* __restrict__ routingOut) {
    __shared__ float xrow[DIN];
    __shared__ float hbuf[NEXP + HID];

    const int b    = blockIdx.x;
    const int tid  = threadIdx.x;
    const int wave = tid >> 5;
    const int lane = tid & 31;

    const float* xr = x + (size_t)b * DIN;
    for (int i = tid; i < DIN; i += 256) xrow[i] = xr[i];
    __syncthreads();

    // 72 dot products of length 1024, one per wave round-robin
    for (int j = wave; j < NEXP + HID; j += 8) {
        const float* wrow = (j < NEXP) ? (gateW + (size_t)j * DIN)
                                       : (capW1 + (size_t)(j - NEXP) * DIN);
        float s = 0.0f;
        for (int k = lane; k < DIN; k += 32) s += xrow[k] * wrow[k];
        #pragma unroll
        for (int off = 16; off > 0; off >>= 1) s += __shfl_down(s, off);
        if (lane == 0) hbuf[j] = (j < NEXP) ? s : (s + capB1[j - NEXP]);
    }
    __syncthreads();

    if (wave == 0) {
        // LayerNorm over hbuf[8..71] (64 values, 2 per lane)
        float h0 = hbuf[NEXP + lane];
        float h1 = hbuf[NEXP + 32 + lane];
        float s  = h0 + h1;
        float s2 = h0 * h0 + h1 * h1;
        #pragma unroll
        for (int off = 16; off > 0; off >>= 1) {
            s  += __shfl_down(s, off);
            s2 += __shfl_down(s2, off);
        }
        s  = __shfl(s, 0);
        s2 = __shfl(s2, 0);
        float mean = s * (1.0f / 64.0f);
        float var  = s2 * (1.0f / 64.0f) - mean * mean;
        float rstd = rsqrtf(var + 1e-5f);
        float g0 = gelu_tanh((h0 - mean) * rstd * lnG[lane]      + lnB[lane]);
        float g1 = gelu_tanh((h1 - mean) * rstd * lnG[lane + 32] + lnB[lane + 32]);
        float c = g0 * capW2[lane] + g1 * capW2[lane + 32];
        #pragma unroll
        for (int off = 16; off > 0; off >>= 1) c += __shfl_down(c, off);
        if (lane == 0) {
            float cap = 1.0f / (1.0f + __expf(-(c + capB2[0])));
            float t = fmaxf(temperature[0], 0.1f);
            float gated[NEXP];
            #pragma unroll
            for (int i = 0; i < NEXP; ++i) gated[i] = hbuf[i] * cap * t;
            int i1 = 0;
            #pragma unroll
            for (int i = 1; i < NEXP; ++i) if (gated[i] > gated[i1]) i1 = i;
            int i2 = (i1 == 0) ? 1 : 0;
            #pragma unroll
            for (int i = 0; i < NEXP; ++i)
                if (i != i1 && gated[i] > gated[i2]) i2 = i;
            float v1 = gated[i1], v2 = gated[i2];
            float rden = 1.0f / (v1 + v2 + 1e-6f);
            float* ro = routingOut + (size_t)b * NEXP;
            #pragma unroll
            for (int i = 0; i < NEXP; ++i)
                ro[i] = (i == i1) ? v1 * rden : ((i == i2) ? v2 * rden : 0.0f);
        }
    }
}

// ---------------------------------------------------------------------------
// Kernel 3/5: bf16 WMMA GEMM  C[m,n] = sum_k A[m,k] * B[n,k]
// MODE 0: B indexed by expert (blockIdx.z); accumulate sum(|C|) -> alignSum[z]
// MODE 1: store C (f32) to Cout
// 256 threads = 8 waves; 128x128 C tile; K-tile 64 double-buffered in LDS via
// global_load_async_to_lds_b128 (ASYNCcnt software pipeline): tile t+1 streams
// into buffer t^1 while tile t feeds the WMMAs.
// ---------------------------------------------------------------------------
template <int MODE>
__global__ void __launch_bounds__(256)
gemm_bf16_kernel(const unsigned short* __restrict__ Abf,
                 const unsigned short* __restrict__ Bbf,
                 float* __restrict__ Cout,
                 float* __restrict__ alignSum,
                 int M, int N, int K) {
    __shared__ unsigned short sA[2][TILE_M * LDS_STRIDE];
    __shared__ unsigned short sB[2][TILE_N * LDS_STRIDE];
    __shared__ float red[256];

    const int tid  = threadIdx.x;
    const int lane = tid & 31;
    const int wave = tid >> 5;
    const int wr   = wave & 1;   // 0..1 : 64-row half
    const int wc   = wave >> 1;  // 0..3 : 32-col strip
    const int ml   = lane & 15;
    const int half = lane >> 4;

    const int rowBase = blockIdx.y * TILE_M;
    const int colBase = blockIdx.x * TILE_N;
    const unsigned short* Bexp =
        (MODE == 0) ? (Bbf + (size_t)blockIdx.z * (size_t)N * (size_t)K) : Bbf;

    // Staging: 128 rows x 64 el = 1024 b128 units per matrix; 4 units/thread.
    const int srow = tid >> 3;        // base row for it=0 (row = srow + it*32)
    const int sseg = tid & 7;         // 8 units (of 8 bf16) per row
    // wait, unit id u = tid + it*256 -> row = u>>3, seg = u&7; since
    // (tid + it*256)>>3 = (tid>>3) + it*32 and seg invariant, precompute:

    // Per-thread issue of one K-tile into LDS buffer `buf` (8 async loads/wave)
    auto issue_tile = [&](int k0, int buf) {
        #pragma unroll
        for (int it = 0; it < 4; ++it) {
            const int row = srow + it * 32;
            const int col = sseg * 8;
            const unsigned short* ga =
                Abf + (size_t)(rowBase + row) * K + k0 + col;
            unsigned la = (unsigned)(unsigned long long)
                (const void*)&sA[buf][row * LDS_STRIDE + col];
            async_load_b128(la, ga);
            const unsigned short* gb =
                Bexp + (size_t)(colBase + row) * K + k0 + col;
            unsigned lb = (unsigned)(unsigned long long)
                (const void*)&sB[buf][row * LDS_STRIDE + col];
            async_load_b128(lb, gb);
        }
    };

    f32x8 acc[4][2];
    #pragma unroll
    for (int r = 0; r < 4; ++r)
        #pragma unroll
        for (int c = 0; c < 2; ++c) acc[r][c] = (f32x8){};

    const int ntile = K / TILE_K;
    issue_tile(0, 0);

    for (int t = 0; t < ntile; ++t) {
        const int buf = t & 1;
        if (t + 1 < ntile) {
            issue_tile((t + 1) * TILE_K, buf ^ 1);
            // <=8 outstanding => the 8 loads of tile t are complete (in-order)
            asm volatile("s_wait_asynccnt 0x8" ::: "memory");
        } else {
            asm volatile("s_wait_asynccnt 0x0" ::: "memory");
        }
        __syncthreads();

        #pragma unroll
        for (int kk = 0; kk < 2; ++kk) {
            // A fragment (16x32 bf16): half h: e0..7 -> K=8h.., e8..15 -> K=16+8h..
            FragAB fa[4];
            #pragma unroll
            for (int r = 0; r < 4; ++r) {
                const unsigned short* base =
                    &sA[buf][(wr * 64 + r * 16 + ml) * LDS_STRIDE + kk * 32];
                fa[r].u[0] = *(const uint4*)(base + 8 * half);
                fa[r].u[1] = *(const uint4*)(base + 16 + 8 * half);
            }
            // B fragment (32x16 bf16): lane n=ml, half h: e0..7 -> K=8h.., e8..15 -> K=16+8h..
            FragAB fb[2];
            #pragma unroll
            for (int c = 0; c < 2; ++c) {
                const unsigned short* base =
                    &sB[buf][(wc * 32 + c * 16 + ml) * LDS_STRIDE + kk * 32 +
                             8 * half];
                fb[c].u[0] = *(const uint4*)(base);
                fb[c].u[1] = *(const uint4*)(base + 16);
            }
            #pragma unroll
            for (int r = 0; r < 4; ++r)
                #pragma unroll
                for (int c = 0; c < 2; ++c)
                    acc[r][c] = __builtin_amdgcn_wmma_f32_16x16x32_bf16(
                        false, fa[r].v, false, fb[c].v,
                        (short)0, acc[r][c], false, false);
        }
        __syncthreads();  // readers done before next issue overwrites buffer
    }

    if (MODE == 0) {
        float s = 0.0f;
        #pragma unroll
        for (int r = 0; r < 4; ++r)
            #pragma unroll
            for (int c = 0; c < 2; ++c)
                #pragma unroll
                for (int i = 0; i < 8; ++i) s += fabsf(acc[r][c][i]);
        red[tid] = s;
        __syncthreads();
        #pragma unroll
        for (int off = 128; off > 0; off >>= 1) {
            if (tid < off) red[tid] += red[tid + off];
            __syncthreads();
        }
        if (tid == 0) atomicAdd(&alignSum[blockIdx.z], red[0]);
    } else {
        // C layout: VGPR i holds (m = i + 8*half, n = ml) within 16x16 subtile
        #pragma unroll
        for (int r = 0; r < 4; ++r)
            #pragma unroll
            for (int c = 0; c < 2; ++c)
                #pragma unroll
                for (int i = 0; i < 8; ++i) {
                    int m = rowBase + wr * 64 + r * 16 + half * 8 + i;
                    int n = colBase + wc * 32 + c * 16 + ml;
                    Cout[(size_t)m * N + n] = acc[r][c][i];
                }
    }
}

// ---------------------------------------------------------------------------
// Kernel 4: w = softmax(alignSum / (B*DOUT)); W_comb = sum_n w[n]*nano_W[n] (bf16)
// ---------------------------------------------------------------------------
__global__ void __launch_bounds__(256)
combine_kernel(const float* __restrict__ nanoW, const float* __restrict__ alignSum,
               unsigned short* __restrict__ wcomb) {
    int i = blockIdx.x * 256 + threadIdx.x;
    if (i >= DOUT * DIN) return;
    const float inv = 1.0f / ((float)BB * (float)DOUT);
    float a[NEXP];
    float mx = -1e30f;
    #pragma unroll
    for (int n = 0; n < NEXP; ++n) {
        a[n] = alignSum[n] * inv;
        mx = fmaxf(mx, a[n]);
    }
    float e[NEXP];
    float den = 0.0f;
    #pragma unroll
    for (int n = 0; n < NEXP; ++n) { e[n] = __expf(a[n] - mx); den += e[n]; }
    float rden = 1.0f / den;
    float acc = 0.0f;
    #pragma unroll
    for (int n = 0; n < NEXP; ++n)
        acc += e[n] * rden * nanoW[(size_t)n * (DOUT * DIN) + i];
    wcomb[i] = f32_to_bf16_rne(acc);
}

// ---------------------------------------------------------------------------
// Launch
// ---------------------------------------------------------------------------
extern "C" void kernel_launch(void* const* d_in, const int* in_sizes, int n_in,
                              void* d_out, int out_size, void* d_ws, size_t ws_size,
                              hipStream_t stream) {
    const float* x     = (const float*)d_in[0];
    const float* gateW = (const float*)d_in[1];
    const float* capW1 = (const float*)d_in[2];
    const float* capB1 = (const float*)d_in[3];
    const float* lnG   = (const float*)d_in[4];
    const float* lnB   = (const float*)d_in[5];
    const float* capW2 = (const float*)d_in[6];
    const float* capB2 = (const float*)d_in[7];
    const float* temp  = (const float*)d_in[8];
    // d_in[9] = scale_weights: softmax sums to 1 -> membrane == x, unused
    const float* nanoW = (const float*)d_in[10];

    float* out         = (float*)d_out;
    float* final_out   = out;                                  // [8192,1024]
    float* routing_out = out + (size_t)BB * DOUT;              // [8192,8]
    float* spikes_out  = routing_out + (size_t)BB * NEXP;      // [8192,1024]

    char* ws = (char*)d_ws;
    unsigned short* xbf   = (unsigned short*)(ws);                      // 16 MB
    unsigned short* wbf   = (unsigned short*)(ws + (size_t)16777216);   // 16 MB
    unsigned short* wcomb = (unsigned short*)(ws + (size_t)33554432);   // 2 MB
    float* alignSum       = (float*)(ws + (size_t)35651584);            // 32 B

    hipMemsetAsync(alignSum, 0, NEXP * sizeof(float), stream);

    const int nx = BB * DIN;          // 8388608
    const int nw = NEXP * DOUT * DIN; // 8388608

    prep_kernel<<<nx / 256, 256, 0, stream>>>(x, xbf, spikes_out, nx);
    convw_kernel<<<nw / 256, 256, 0, stream>>>(nanoW, wbf, nw);

    router_kernel<<<BB, 256, 0, stream>>>(x, gateW, capW1, capB1, lnG, lnB,
                                          capW2, capB2, temp, routing_out);

    // Pass 1: 8 expert GEMMs, reduce sum|outs| per expert (no outs in HBM)
    dim3 g0(DOUT / TILE_N, BB / TILE_M, NEXP);
    gemm_bf16_kernel<0><<<g0, 256, 0, stream>>>(xbf, wbf, nullptr, alignSum,
                                                BB, DOUT, DIN);

    // softmax-combined weight matrix
    combine_kernel<<<(DOUT * DIN) / 256, 256, 0, stream>>>(nanoW, alignSum, wcomb);

    // Pass 2: final = x @ W_comb^T
    dim3 g1(DOUT / TILE_N, BB / TILE_M, 1);
    gemm_bf16_kernel<1><<<g1, 256, 0, stream>>>(xbf, wcomb, final_out, nullptr,
                                                BB, DOUT, DIN);
}
